// Transformer_81449759801619
// MI455X (gfx1250) — matrix operand order repaired and checked by reading, not compiled, not run on hardware
//
#include <hip/hip_runtime.h>
#include <hip/hip_bf16.h>

// ---------------------------------------------------------------------------
// Transformer encoder-decoder forward for MI455X (gfx1250, wave32, WMMA).
//
// Compute-bound workload (~155 GFLOP, all data L2-resident at 192MB), so all
// large matmuls (projections, FFN, encoder/cross attention) go through
// v_wmma_f32_16x16x32_f16 (f16 inputs, f32 accumulate). The reference's buggy
// post-softmax causal mask (-1e10 fill) is not f16-representable, so the tiny
// decoder self-attention (100x100 per head) stays in f32 VALU, faithfully.
// ---------------------------------------------------------------------------

typedef __attribute__((ext_vector_type(16))) _Float16 v16h;
typedef __attribute__((ext_vector_type(8)))  float    v8f;

#define DMODEL 512
#define NH     8
#define DHEAD  64
#define DFF    2048
#define BATCH  8
#define TENC   1024
#define TDEC   100

__device__ __forceinline__ v8f wmma16x16x32(v16h a, v16h b, v8f c) {
  return __builtin_amdgcn_wmma_f32_16x16x32_f16(false, a, false, b, (short)0, c,
                                                false, false);
}

// A-fragment (16x32, f16) from a global f32 row-major [rows][stride] source.
// Contiguous k order (not ISA table order): lanes 0-15 row M=lane hold K=0..15; lanes 16-31 hold K=16..31.
__device__ __forceinline__ v16h frag_a_global(const float* base, int stride,
                                              int row0, int maxRow, int kofs) {
  int lane = threadIdx.x & 31;
  int r = row0 + (lane & 15);
  int kb = ((lane >> 4) << 4) + kofs;
  v16h f;
  if (r < maxRow) {
    const float* p = base + (size_t)r * stride + kb;
#pragma unroll
    for (int i = 0; i < 16; ++i) f[i] = (_Float16)p[i];
  } else {
#pragma unroll
    for (int i = 0; i < 16; ++i) f[i] = (_Float16)0.0f;
  }
  return f;
}

// B-fragment (32x16, f16) from a source stored [N][K] row-major (e.g. keys
// [key][dk]).  Contiguous k order, as A: lanes 0-15 col N=lane hold K=0..15 contiguous,
// lanes 16-31 hold K=16..31.
__device__ __forceinline__ v16h frag_b_nk_global(const float* base, int stride,
                                                 int kofs) {
  int lane = threadIdx.x & 31;
  int n = lane & 15;
  int kb = ((lane >> 4) << 4) + kofs;
  const float* p = base + (size_t)n * stride + kb;
  v16h f;
#pragma unroll
  for (int i = 0; i < 16; ++i) f[i] = (_Float16)p[i];
  return f;
}

// ---------------------------------------------------------------------------
// GEMM: C[M,N] = act(A[M,K] @ W[K,N] + bias).  64x64 block tile, 256 threads
// (8 waves: 4 M-tiles x 2 N-tiles of 16x32), K-step 32, f16 staging in LDS.
// ---------------------------------------------------------------------------
__global__ __launch_bounds__(256) void gemm_kernel(
    const float* __restrict__ A, const float* __restrict__ W,
    const float* __restrict__ bias, float* __restrict__ C,
    int M, int N, int K, int relu)
{
  __shared__ _Float16 sA[64 * 40];   // [row][k], pitch 40 halves
  __shared__ _Float16 sB[64 * 40];   // [n][k] (W tile transposed), pitch 40

  const int tid  = threadIdx.x;
  const int lane = tid & 31;
  const int wid  = tid >> 5;
  const int wm   = wid & 3;          // 0..3 -> M sub-tile
  const int wn   = wid >> 2;         // 0..1 -> N sub-tile (32 wide)
  const int m0 = blockIdx.y * 64, n0 = blockIdx.x * 64;

  const int arow = tid >> 2, acol = (tid & 3) * 8;   // A staging: 64 rows x 32k
  const int brow = tid >> 3, bcol = (tid & 7) * 8;   // W staging: 32 k x 64 n

  v8f acc0 = {}; v8f acc1 = {};

  for (int k0 = 0; k0 < K; k0 += 32) {
    {
      int gr = m0 + arow;
      if (gr < M) {
        const float* p = A + (size_t)gr * K + k0 + acol;
#pragma unroll
        for (int i = 0; i < 8; ++i) sA[arow * 40 + acol + i] = (_Float16)p[i];
        if (k0 + 32 < K) __builtin_prefetch(p + 32, 0, 1);  // global_prefetch_b8
      } else {
#pragma unroll
        for (int i = 0; i < 8; ++i) sA[arow * 40 + acol + i] = (_Float16)0.0f;
      }
    }
    {
      const float* p = W + (size_t)(k0 + brow) * N + n0 + bcol;
#pragma unroll
      for (int i = 0; i < 8; ++i) sB[(bcol + i) * 40 + brow] = (_Float16)p[i];
    }
    __syncthreads();

    v16h a;
    {
      int r = lane & 15, kb = (lane >> 4) << 4;
      const _Float16* p = sA + (wm * 16 + r) * 40 + kb;
#pragma unroll
      for (int i = 0; i < 16; ++i) a[i] = p[i];
    }
    v16h b0, b1;
    {
      int n = lane & 15, kb = (lane >> 4) << 4;
      const _Float16* p0 = sB + (wn * 32 + n) * 40 + kb;
      const _Float16* p1 = sB + (wn * 32 + 16 + n) * 40 + kb;
#pragma unroll
      for (int i = 0; i < 16; ++i) { b0[i] = p0[i]; b1[i] = p1[i]; }
    }
    acc0 = wmma16x16x32(a, b0, acc0);
    acc1 = wmma16x16x32(a, b1, acc1);
    __syncthreads();
  }

  // Epilogue.  C/D layout: VGPR r -> row r (lanes 0-15) / r+8 (lanes 16-31),
  // col = lane&15.
  const int rbase = m0 + wm * 16 + ((lane >> 4) << 3);
#pragma unroll
  for (int nt = 0; nt < 2; ++nt) {
    v8f acc = nt ? acc1 : acc0;
    int col = n0 + wn * 32 + nt * 16 + (lane & 15);
    float bv = bias ? bias[col] : 0.0f;
#pragma unroll
    for (int r = 0; r < 8; ++r) {
      int row = rbase + r;
      if (row < M) {
        float v = acc[r] + bv;
        if (relu) v = fmaxf(v, 0.0f);
        C[(size_t)row * N + col] = v;
      }
    }
  }
}

// ---------------------------------------------------------------------------
// Non-causal attention (encoder self + decoder cross).  One workgroup per
// (b, h, 16-query tile); 4 waves.  Scores in f32 dynamic LDS (16 x Lk = 64KB),
// wave-shuffle softmax, WMMA for QK^T and PV.  Lk % 32 == 0 (always 1024 here).
// ---------------------------------------------------------------------------
__global__ __launch_bounds__(128) void attn_kernel(
    const float* __restrict__ Q, const float* __restrict__ K,
    const float* __restrict__ V, float* __restrict__ O,
    int Lq, int Lk)
{
  extern __shared__ float sS[];                 // 16 * Lk floats
  const int lane = threadIdx.x & 31;
  const int w    = threadIdx.x >> 5;            // wave 0..3
  const int h = blockIdx.y, b = blockIdx.z;
  const int q0 = blockIdx.x * 16;

  const float* Qb = Q + (size_t)b * Lq * DMODEL + h * DHEAD;
  const float* Kb = K + (size_t)b * Lk * DMODEL + h * DHEAD;
  const float* Vb = V + (size_t)b * Lk * DMODEL + h * DHEAD;

  v16h aq0 = frag_a_global(Qb, DMODEL, q0, Lq, 0);
  v16h aq1 = frag_a_global(Qb, DMODEL, q0, Lq, 32);

  // ---- scores = (q @ k^T) * rsqrt(64); waves split key tiles ----
  const int nkt = Lk >> 4;
  for (int kt = w; kt < nkt; kt += 4) {
    const float* kp = Kb + (size_t)kt * 16 * DMODEL;
    v16h b0 = frag_b_nk_global(kp, DMODEL, 0);
    v16h b1 = frag_b_nk_global(kp, DMODEL, 32);
    v8f s = {};
    s = wmma16x16x32(aq0, b0, s);
    s = wmma16x16x32(aq1, b1, s);
    int col = kt * 16 + (lane & 15);
    int rb = (lane >> 4) << 3;
#pragma unroll
    for (int r = 0; r < 8; ++r) sS[(rb + r) * Lk + col] = s[r] * 0.125f;
  }
  __syncthreads();

  // ---- row softmax: wave w handles rows 4w..4w+3, shuffle reductions ----
  for (int r = w * 4; r < w * 4 + 4; ++r) {
    float* row = sS + (size_t)r * Lk;
    float m = -3.402823e38f;
    for (int k2 = lane; k2 < Lk; k2 += 32) m = fmaxf(m, row[k2]);
#pragma unroll
    for (int off = 16; off > 0; off >>= 1) m = fmaxf(m, __shfl_xor(m, off, 32));
    float sum = 0.0f;
    for (int k2 = lane; k2 < Lk; k2 += 32) {
      float e = __expf(row[k2] - m); row[k2] = e; sum += e;
    }
#pragma unroll
    for (int off = 16; off > 0; off >>= 1) sum += __shfl_xor(sum, off, 32);
    float inv = 1.0f / sum;
    for (int k2 = lane; k2 < Lk; k2 += 32) row[k2] *= inv;
  }
  __syncthreads();

  // ---- o = p @ v; wave w owns dv slice [w*16, w*16+16) ----
  v8f o = {};
  for (int kc = 0; kc < (Lk >> 5); ++kc) {
    v16h ap;
    {
      int r = lane & 15, kb = (lane >> 4) << 4;
      const float* p = sS + (size_t)r * Lk + kc * 32 + kb;
#pragma unroll
      for (int i = 0; i < 16; ++i) ap[i] = (_Float16)p[i];
    }
    v16h bv;
    {
      int n = lane & 15, kb = (lane >> 4) << 4;
      const float* p = Vb + (size_t)(kc * 32 + kb) * DMODEL + w * 16 + n;
#pragma unroll
      for (int i = 0; i < 16; ++i) bv[i] = (_Float16)p[(size_t)i * DMODEL];
    }
    o = wmma16x16x32(ap, bv, o);
  }
  float* Ob = O + (size_t)b * Lq * DMODEL + h * DHEAD + w * 16;
  int col = lane & 15;
  int rb = (lane >> 4) << 3;
#pragma unroll
  for (int r = 0; r < 8; ++r) {
    int gr = q0 + rb + r;
    if (gr < Lq) Ob[(size_t)gr * DMODEL + col] = o[r];
  }
}

// ---------------------------------------------------------------------------
// Decoder causal self-attention, f32, faithful to the reference's buggy
// post-softmax -1e10 mask fill (not f16-representable).  100x100 per (b,h),
// negligible FLOPs.  One workgroup per (b,h).
// ---------------------------------------------------------------------------
__global__ __launch_bounds__(256) void causal_attn_kernel(
    const float* __restrict__ Q, const float* __restrict__ K,
    const float* __restrict__ V, float* __restrict__ O)
{
  __shared__ float sS[TDEC * TDEC];            // 40000 B
  const int b = blockIdx.x >> 3;
  const int h = blockIdx.x & 7;
  const int t = threadIdx.x;
  const float* Qb = Q + (size_t)b * TDEC * DMODEL + h * DHEAD;
  const float* Kb = K + (size_t)b * TDEC * DMODEL + h * DHEAD;
  const float* Vb = V + (size_t)b * TDEC * DMODEL + h * DHEAD;
  float* Ob = O + (size_t)b * TDEC * DMODEL + h * DHEAD;

  for (int idx = t; idx < TDEC * TDEC; idx += 256) {
    int q = idx / TDEC, k = idx % TDEC;
    const float* qp = Qb + (size_t)q * DMODEL;
    const float* kp = Kb + (size_t)k * DMODEL;
    float d = 0.0f;
#pragma unroll 8
    for (int e = 0; e < DHEAD; ++e) d += qp[e] * kp[e];
    sS[idx] = d * 0.125f;
  }
  __syncthreads();
  if (t < TDEC) {
    float* row = sS + t * TDEC;
    float m = -3.402823e38f;
    for (int k = 0; k < TDEC; ++k) m = fmaxf(m, row[k]);
    float sum = 0.0f;
    for (int k = 0; k < TDEC; ++k) { float e = __expf(row[k] - m); row[k] = e; sum += e; }
    float inv = 1.0f / sum;
    for (int k = 0; k < TDEC; ++k) row[k] = (k <= t) ? row[k] * inv : -1e10f;
  }
  __syncthreads();
  for (int idx = t; idx < TDEC * DHEAD; idx += 256) {
    int q = idx / DHEAD, d = idx % DHEAD;
    const float* pr = sS + q * TDEC;
    float acc = 0.0f;
    for (int k = 0; k < TDEC; ++k) acc += pr[k] * Vb[(size_t)k * DMODEL + d];
    Ob[(size_t)q * DMODEL + d] = acc;
  }
}

// ---------------------------------------------------------------------------
// out[row] = LayerNorm(A[row] + B[row]) * g + beta      (d_model = 512)
// ---------------------------------------------------------------------------
__global__ __launch_bounds__(128) void add_ln_kernel(
    const float* __restrict__ A, const float* __restrict__ Bs,
    const float* __restrict__ g, const float* __restrict__ beta,
    float* __restrict__ out)
{
  __shared__ float r1[128], r2[128];
  const int t = threadIdx.x;
  const size_t row = blockIdx.x;
  const float* a  = A  + row * DMODEL;
  const float* bb = Bs + row * DMODEL;
  float v[4]; float s = 0.0f, ss = 0.0f;
#pragma unroll
  for (int i = 0; i < 4; ++i) {
    int c = t + i * 128;
    v[i] = a[c] + bb[c];
    s += v[i]; ss += v[i] * v[i];
  }
  r1[t] = s; r2[t] = ss;
  __syncthreads();
  for (int off = 64; off > 0; off >>= 1) {
    if (t < off) { r1[t] += r1[t + off]; r2[t] += r2[t + off]; }
    __syncthreads();
  }
  float mean = r1[0] * (1.0f / DMODEL);
  float var  = r2[0] * (1.0f / DMODEL) - mean * mean;
  float rstd = rsqrtf(var + 1e-5f);
#pragma unroll
  for (int i = 0; i < 4; ++i) {
    int c = t + i * 128;
    out[row * DMODEL + c] = (v[i] - mean) * rstd * g[c] + beta[c];
  }
}

// ---------------------------------------------------------------------------
// Host orchestration.
// ---------------------------------------------------------------------------
extern "C" void kernel_launch(void* const* d_in, const int* in_sizes, int n_in,
                              void* d_out, int out_size, void* d_ws, size_t ws_size,
                              hipStream_t stream)
{
  const float* x        = (const float*)d_in[0];
  const float* y        = (const float*)d_in[1];
  const float* enc_Wq   = (const float*)d_in[2];
  const float* enc_Wk   = (const float*)d_in[3];
  const float* enc_Wv   = (const float*)d_in[4];
  const float* enc_w1   = (const float*)d_in[5];
  const float* enc_b1   = (const float*)d_in[6];
  const float* enc_w2   = (const float*)d_in[7];
  const float* enc_b2   = (const float*)d_in[8];
  const float* enc_ln1g = (const float*)d_in[9];
  const float* enc_ln1b = (const float*)d_in[10];
  const float* enc_ln2g = (const float*)d_in[11];
  const float* enc_ln2b = (const float*)d_in[12];
  const float* dec_Wq1  = (const float*)d_in[13];
  const float* dec_Wk1  = (const float*)d_in[14];
  const float* dec_Wv1  = (const float*)d_in[15];
  const float* dec_Wq2  = (const float*)d_in[16];
  const float* dec_Wk2  = (const float*)d_in[17];
  const float* dec_Wv2  = (const float*)d_in[18];
  const float* dec_w1   = (const float*)d_in[19];
  const float* dec_b1   = (const float*)d_in[20];
  const float* dec_w2   = (const float*)d_in[21];
  const float* dec_b2   = (const float*)d_in[22];
  const float* dec_ln1g = (const float*)d_in[23];
  const float* dec_ln1b = (const float*)d_in[24];
  const float* dec_ln2g = (const float*)d_in[25];
  const float* dec_ln2b = (const float*)d_in[26];
  const float* dec_ln3g = (const float*)d_in[27];
  const float* dec_ln3b = (const float*)d_in[28];

  // Workspace layout (floats): 6 slots of B*TENC*DMODEL + 1 FF slot (4x).
  const size_t Nel = (size_t)BATCH * TENC * DMODEL;   // 4,194,304 floats
  float* X  = (float*)d_ws;
  float* Yb = X  + Nel;
  float* Qb = Yb + Nel;
  float* Kb = Qb + Nel;
  float* Vb = Kb + Nel;
  float* Ab = Vb + Nel;
  float* FF = Ab + Nel;                               // B*TENC*DFF floats

  (void)ws_size; (void)in_sizes; (void)n_in; (void)out_size;

  hipMemcpyAsync(X,  x, Nel * sizeof(float), hipMemcpyDeviceToDevice, stream);
  hipMemcpyAsync(Yb, y, (size_t)BATCH * TDEC * DMODEL * sizeof(float),
                 hipMemcpyDeviceToDevice, stream);

  // 16 rows x 1024 keys x f32 = 64KB dynamic LDS (well under 320KB/WGP).
  (void)hipFuncSetAttribute(reinterpret_cast<const void*>(attn_kernel),
                            hipFuncAttributeMaxDynamicSharedMemorySize,
                            16 * TENC * (int)sizeof(float));
  const unsigned attnShm = 16 * TENC * sizeof(float);

  const int Me = BATCH * TENC;   // 8192 encoder tokens
  const int Md = BATCH * TDEC;   // 800 decoder tokens
  auto gg = [](int M, int N) { return dim3((unsigned)(N / 64), (unsigned)((M + 63) / 64)); };

  // ---------------- Encoder ----------------
  for (int i = 0; i < 2; ++i) {
    const float* Wq = enc_Wq + (size_t)i * DMODEL * DMODEL;
    const float* Wk = enc_Wk + (size_t)i * DMODEL * DMODEL;
    const float* Wv = enc_Wv + (size_t)i * DMODEL * DMODEL;
    gemm_kernel<<<gg(Me, DMODEL), 256, 0, stream>>>(X, Wq, nullptr, Qb, Me, DMODEL, DMODEL, 0);
    gemm_kernel<<<gg(Me, DMODEL), 256, 0, stream>>>(X, Wk, nullptr, Kb, Me, DMODEL, DMODEL, 0);
    gemm_kernel<<<gg(Me, DMODEL), 256, 0, stream>>>(X, Wv, nullptr, Vb, Me, DMODEL, DMODEL, 0);
    attn_kernel<<<dim3(TENC / 16, NH, BATCH), 128, attnShm, stream>>>(Qb, Kb, Vb, Ab, TENC, TENC);
    add_ln_kernel<<<Me, 128, 0, stream>>>(X, Ab, enc_ln1g + i * DMODEL, enc_ln1b + i * DMODEL, Qb); // X1 -> Qb
    gemm_kernel<<<gg(Me, DFF), 256, 0, stream>>>(Qb, enc_w1 + (size_t)i * DMODEL * DFF,
                                                 enc_b1 + (size_t)i * DFF, FF, Me, DFF, DMODEL, 1);
    gemm_kernel<<<gg(Me, DMODEL), 256, 0, stream>>>(FF, enc_w2 + (size_t)i * DFF * DMODEL,
                                                    enc_b2 + (size_t)i * DMODEL, Kb, Me, DMODEL, DFF, 0);
    add_ln_kernel<<<Me, 128, 0, stream>>>(Qb, Kb, enc_ln2g + i * DMODEL, enc_ln2b + i * DMODEL, X);
  }

  // ---------------- Decoder ----------------
  for (int i = 0; i < 2; ++i) {
    const float* Wq1 = dec_Wq1 + (size_t)i * DMODEL * DMODEL;
    const float* Wk1 = dec_Wk1 + (size_t)i * DMODEL * DMODEL;
    const float* Wv1 = dec_Wv1 + (size_t)i * DMODEL * DMODEL;
    const float* Wq2 = dec_Wq2 + (size_t)i * DMODEL * DMODEL;
    const float* Wk2 = dec_Wk2 + (size_t)i * DMODEL * DMODEL;
    const float* Wv2 = dec_Wv2 + (size_t)i * DMODEL * DMODEL;

    // causal self-attention
    gemm_kernel<<<gg(Md, DMODEL), 256, 0, stream>>>(Yb, Wq1, nullptr, Qb, Md, DMODEL, DMODEL, 0);
    gemm_kernel<<<gg(Md, DMODEL), 256, 0, stream>>>(Yb, Wk1, nullptr, Kb, Md, DMODEL, DMODEL, 0);
    gemm_kernel<<<gg(Md, DMODEL), 256, 0, stream>>>(Yb, Wv1, nullptr, Vb, Md, DMODEL, DMODEL, 0);
    causal_attn_kernel<<<BATCH * NH, 256, 0, stream>>>(Qb, Kb, Vb, Ab);
    add_ln_kernel<<<Md, 128, 0, stream>>>(Yb, Ab, dec_ln1g + i * DMODEL, dec_ln1b + i * DMODEL, Qb); // Y1 -> Qb

    // cross-attention against encoder memory X
    gemm_kernel<<<gg(Md, DMODEL), 256, 0, stream>>>(Qb, Wq2, nullptr, Ab, Md, DMODEL, DMODEL, 0);    // Q2 -> Ab
    gemm_kernel<<<gg(Me, DMODEL), 256, 0, stream>>>(X, Wk2, nullptr, Kb, Me, DMODEL, DMODEL, 0);     // Kx
    gemm_kernel<<<gg(Me, DMODEL), 256, 0, stream>>>(X, Wv2, nullptr, Vb, Me, DMODEL, DMODEL, 0);     // Vx
    attn_kernel<<<dim3((TDEC + 15) / 16, NH, BATCH), 128, attnShm, stream>>>(Ab, Kb, Vb, Yb, TDEC, TENC); // A2 -> Yb
    add_ln_kernel<<<Md, 128, 0, stream>>>(Qb, Yb, dec_ln2g + i * DMODEL, dec_ln2b + i * DMODEL, Ab); // Y2 -> Ab

    // FFN
    gemm_kernel<<<gg(Md, DFF), 256, 0, stream>>>(Ab, dec_w1 + (size_t)i * DMODEL * DFF,
                                                 dec_b1 + (size_t)i * DFF, FF, Md, DFF, DMODEL, 1);
    gemm_kernel<<<gg(Md, DMODEL), 256, 0, stream>>>(FF, dec_w2 + (size_t)i * DFF * DMODEL,
                                                    dec_b2 + (size_t)i * DMODEL, Vb, Md, DMODEL, DFF, 0);
    float* outp = (i == 1) ? (float*)d_out : Yb;
    add_ln_kernel<<<Md, 128, 0, stream>>>(Ab, Vb, dec_ln3g + i * DMODEL, dec_ln3b + i * DMODEL, outp);
  }
}
